// SelfAttention_49151605735824
// MI455X (gfx1250) — matrix operand order repaired and checked by reading, not compiled, not run on hardware
//
#include <hip/hip_runtime.h>
#include <hip/hip_bf16.h>
#include <math.h>

typedef __attribute__((ext_vector_type(16))) _Float16 v16h;
typedef __attribute__((ext_vector_type(8)))  _Float16 v8h;
typedef __attribute__((ext_vector_type(8)))  float    v8f;
typedef unsigned int u32x4 __attribute__((ext_vector_type(4)));
typedef int          i32x4 __attribute__((ext_vector_type(4)));
typedef int          i32x8 __attribute__((ext_vector_type(8)));

#define BDIM 2
#define SEQ  2048
#define DIM  1024
#define NH   16
#define HD   64
#define TD   3072   // 3*DIM

union H16 { v16h v; v8h h8[2]; _Float16 h[16]; };
union F8  { v8f v; float f[8]; };

#define WMMA_F16(a,b,c) __builtin_amdgcn_wmma_f32_16x16x32_f16(false,(a),false,(b),(short)0,(c),false,false)

#if __has_builtin(__builtin_amdgcn_tensor_load_to_lds)
#define HAVE_TDM 1
#endif

// ---- fragment loaders (contiguous k order, same in A and B) ------------------------------
// A (16x32 f16, MxK): lane L holds row L&15; lanes<16: K = k0+0..15
//                     lanes>=16: K = k0+16..31
static __device__ inline v16h load_a_frag(const _Float16* A, int ld, int r0, int k0, int lane) {
  int row = r0 + (lane & 15);
  int kb  = k0 + ((lane >> 4) << 4);
  H16 u;
  u.h8[0] = *(const v8h*)(A + (size_t)row * ld + kb);
  u.h8[1] = *(const v8h*)(A + (size_t)row * ld + kb + 8);
  return u.v;
}
// B (32x16 f16, KxN) sourced from Bt stored as (N rows x Kdim cols) row-major:
// lane L holds column n0+(L&15); lanes<16 hold K=k0..k0+15, lanes>=16 K=k0+16..k0+31.
static __device__ inline v16h load_b_frag(const _Float16* Bt, int ld, int n0, int k0, int lane) {
  int c    = n0 + (lane & 15);
  int koff = k0 + ((lane >> 4) << 4);
  return *(const v16h*)(Bt + (size_t)c * ld + koff);
}

// ---- Tensor Data Mover: 2D f16 tile -> LDS (ISA ch8 D# layout) ----------
static __device__ inline void tdm_wait0() {
#if __has_builtin(__builtin_amdgcn_s_wait_tensorcnt)
  __builtin_amdgcn_s_wait_tensorcnt(0);
#else
  asm volatile("s_wait_tensorcnt 0x0" ::: "memory");
#endif
}
#ifdef HAVE_TDM
static __device__ inline void tdm_load_tile_f16(const _Float16* gsrc, unsigned lds_byte_off,
                                                int tileW, int tileH, int strideElems) {
  unsigned long long ga = (unsigned long long)(const void*)gsrc;
  u32x4 g0;
  g0[0] = 1u;                                         // count=1 (valid user descriptor)
  g0[1] = lds_byte_off;                               // lds_addr
  g0[2] = (unsigned)(ga & 0xFFFFFFFFu);               // global_addr[31:0]
  g0[3] = (unsigned)((ga >> 32) & 0x01FFFFFFu) | 0x80000000u; // addr[56:32] | type=2
  i32x8 g1;
  g1[0] = 0x00010000;                                 // wg_mask=0 (no cluster), data_size=1 (2B)
  g1[1] = tileW << 16;                                // tensor_dim0[15:0]  (bits 63:48)
  g1[2] = tileH << 16;                                // tensor_dim1[15:0]  (bits 95:80)
  g1[3] = tileW << 16;                                // tile_dim0          (bits 127:112)
  g1[4] = tileH;                                      // tile_dim1; tile_dim2=0
  g1[5] = strideElems;                                // tensor_dim0_stride[31:0]
  g1[6] = 0;                                          // stride[47:32] | dim1_stride lo (unused, 2D)
  g1[7] = 0;
  i32x4 z4 = {0, 0, 0, 0};
#if __clang_major__ >= 23
  i32x8 z8 = {0, 0, 0, 0, 0, 0, 0, 0};
  __builtin_amdgcn_tensor_load_to_lds(g0, g1, z4, z4, z8, 0);
#else
  __builtin_amdgcn_tensor_load_to_lds(g0, g1, z4, z4, 0);
#endif
}
#endif

// ---- precision prep kernels ---------------------------------------------
__global__ void __launch_bounds__(256) k_transpose_x(const float* __restrict__ x,
                                                     _Float16* __restrict__ Xh) {
  int idx = blockIdx.x * 256 + threadIdx.x;          // over B*S*D, Xh row-major (b,s,d)
  int bi = idx / (SEQ * DIM);
  int r  = idx - bi * SEQ * DIM;
  int sidx = r / DIM;
  int dcol = r - sidx * DIM;
  Xh[idx] = (_Float16)x[(size_t)bi * DIM * SEQ + (size_t)dcol * SEQ + sidx];
}
__global__ void __launch_bounds__(256) k_w1t(const float* __restrict__ W1,
                                             _Float16* __restrict__ W1t) {
  int idx = blockIdx.x * 256 + threadIdx.x;          // W1t: (TD rows x DIM cols)
  int n = idx / DIM, k = idx - n * DIM;
  W1t[idx] = (_Float16)W1[(size_t)k * TD + n];
}
__global__ void __launch_bounds__(256) k_w2t(const float* __restrict__ W2,
                                             _Float16* __restrict__ W2t) {
  int idx = blockIdx.x * 256 + threadIdx.x;          // W2t: (DIM rows x DIM cols)
  int n = idx / DIM, k = idx - n * DIM;
  W2t[idx] = (_Float16)W2[(size_t)k * DIM + n];
}

// ---- QKV projection: (B*S, DIM) x (DIM, 3*DIM) --------------------------
// Block = 4 waves. Block tile 32 x 256; wave tile 32 x 64 (2x4 WMMA frags).
// Shared 32x32 A tile staged into LDS by the Tensor Data Mover.
__global__ void __launch_bounds__(128) k_qkv(const _Float16* __restrict__ Xh,
                                             const _Float16* __restrict__ W1t,
                                             _Float16* __restrict__ Q,
                                             _Float16* __restrict__ K,
                                             _Float16* __restrict__ Vt) {
  __shared__ __align__(128) _Float16 tileA[32 * 32];
  int lane = threadIdx.x & 31;
  int wid  = threadIdx.x >> 5;
  int rt = blockIdx.x / (TD / 256);                  // 128 row-tiles x 12 col-tiles
  int ct = blockIdx.x - rt * (TD / 256);
  int g0r = rt * 32;
  int n0  = ct * 256 + wid * 64;
  v8f c[2][4];
  #pragma unroll
  for (int i = 0; i < 2; i++)
    #pragma unroll
    for (int j = 0; j < 4; j++) c[i][j] = (v8f){};
  unsigned laddrA = (unsigned)(unsigned long long)(void*)tileA; // flat low 32b == LDS offset

  for (int k0 = 0; k0 < DIM; k0 += 32) {
#ifdef HAVE_TDM
    if (wid == 0) {
      tdm_load_tile_f16(Xh + (size_t)g0r * DIM + k0, laddrA, 32, 32, DIM);
      tdm_wait0();
    }
#else
    { int row = threadIdx.x >> 2, kc = (threadIdx.x & 3) * 8;
      *(v8h*)(tileA + row * 32 + kc) = *(const v8h*)(Xh + (size_t)(g0r + row) * DIM + k0 + kc); }
#endif
    __syncthreads();
    v16h a0 = load_a_frag(tileA, 32, 0,  0, lane);
    v16h a1 = load_a_frag(tileA, 32, 16, 0, lane);
    __builtin_prefetch(W1t + (size_t)(n0 + (lane & 15)) * DIM + k0 + 64, 0, 1);
    #pragma unroll
    for (int j = 0; j < 4; j++) {
      v16h b = load_b_frag(W1t, DIM, n0 + j * 16, k0, lane);
      c[0][j] = WMMA_F16(a0, b, c[0][j]);
      c[1][j] = WMMA_F16(a1, b, c[1][j]);
    }
    __syncthreads();
  }
  int rbase = (lane >> 4) << 3, colL = lane & 15;
  #pragma unroll
  for (int rf = 0; rf < 2; rf++) {
    #pragma unroll
    for (int j = 0; j < 4; j++) {
      F8 cc; cc.v = c[rf][j];
      int n = n0 + j * 16 + colL;
      int which = n >> 10;
      int w = n & 1023;
      int h = w >> 6, hd = w & 63;
      #pragma unroll
      for (int r = 0; r < 8; r++) {
        int g = g0r + rf * 16 + rbase + r;
        int bi = g >> 11, sidx = g & (SEQ - 1);
        _Float16 v = (_Float16)cc.f[r];
        if (which == 0)      Q [(((size_t)(bi * NH + h)) * SEQ + sidx) * HD + hd] = v;
        else if (which == 1) K [(((size_t)(bi * NH + h)) * SEQ + sidx) * HD + hd] = v;
        else                 Vt[(((size_t)(bi * NH + h)) * HD + hd) * SEQ + sidx] = v;
      }
    }
  }
}

// ---- causal flash attention: one wave per 16-row q tile -----------------
__global__ void __launch_bounds__(128) k_attn(const _Float16* __restrict__ Q,
                                              const _Float16* __restrict__ K,
                                              const _Float16* __restrict__ Vt,
                                              _Float16* __restrict__ AO) {
  __shared__ __align__(64) _Float16 pl[4][16 * 32];  // wave-private P staging
  int lane = threadIdx.x & 31;
  int wid  = threadIdx.x >> 5;
  int wg = blockIdx.x * 4 + wid;                     // B*NH*(SEQ/16) = 4096 waves
  int qt = wg & ((SEQ / 16) - 1);
  int hi = (wg >> 7) & (NH - 1);
  int bi = wg >> 11;
  int q0 = qt * 16;
  const _Float16* Qp = Q  + ((size_t)(bi * NH + hi)) * SEQ * HD;
  const _Float16* Kp = K  + ((size_t)(bi * NH + hi)) * SEQ * HD;
  const _Float16* Vp = Vt + ((size_t)(bi * NH + hi)) * HD * SEQ;

  v16h aq0 = load_a_frag(Qp, HD, q0, 0, lane);
  v16h aq1 = load_a_frag(Qp, HD, q0, 32, lane);

  float m[8], l[8];
  F8 o0, o1, o2, o3;
  #pragma unroll
  for (int r = 0; r < 8; r++) {
    m[r] = -INFINITY; l[r] = 0.f;
    o0.f[r] = 0.f; o1.f[r] = 0.f; o2.f[r] = 0.f; o3.f[r] = 0.f;
  }
  const float scale = 0.125f;                        // 1/sqrt(64)
  const float NEGBIG = -3.4028234663852886e38f;      // float32 finfo.min (additive mask)
  int rbase = (lane >> 4) << 3;
  int colL  = lane & 15;
  _Float16* plw = pl[wid];

  int ktmax = (q0 + 15) >> 5;                        // causal bound, 32 keys/iter
  for (int kt = 0; kt <= ktmax; kt++) {
    int kb = kt * 32;
    v8f s0 = {}, s1 = {};
    s0 = WMMA_F16(aq0, load_b_frag(Kp, HD, kb,      0,  lane), s0);
    s0 = WMMA_F16(aq1, load_b_frag(Kp, HD, kb,      32, lane), s0);
    s1 = WMMA_F16(aq0, load_b_frag(Kp, HD, kb + 16, 0,  lane), s1);
    s1 = WMMA_F16(aq1, load_b_frag(Kp, HD, kb + 16, 32, lane), s1);
    F8 f0, f1; f0.v = s0; f1.v = s1;

    float p0[8], p1[8];
    #pragma unroll
    for (int r = 0; r < 8; r++) {
      int row = q0 + rbase + r;
      int c0 = kb + colL, c1 = c0 + 16;
      float v0 = f0.f[r] * scale; if (c0 > row) v0 += NEGBIG;
      float v1 = f1.f[r] * scale; if (c1 > row) v1 += NEGBIG;
      float t = fmaxf(v0, v1);
      #pragma unroll
      for (int off = 1; off < 16; off <<= 1) t = fmaxf(t, __shfl_xor(t, off, 32));
      float nm = fmaxf(m[r], t);
      float alpha = __expf(m[r] - nm);
      float e0 = __expf(v0 - nm), e1 = __expf(v1 - nm);
      float rs = e0 + e1;
      #pragma unroll
      for (int off = 1; off < 16; off <<= 1) rs += __shfl_xor(rs, off, 32);
      l[r] = l[r] * alpha + rs;
      m[r] = nm;
      p0[r] = e0; p1[r] = e1;
      o0.f[r] *= alpha; o1.f[r] *= alpha; o2.f[r] *= alpha; o3.f[r] *= alpha;
    }
    // C-layout -> A-layout transpose via wave-private LDS (no barrier needed)
    #pragma unroll
    for (int r = 0; r < 8; r++) {
      int row = rbase + r;
      plw[row * 32 + colL]      = (_Float16)p0[r];
      plw[row * 32 + colL + 16] = (_Float16)p1[r];
    }
    v16h ap = load_a_frag(plw, 32, 0, 0, lane);
    o0.v = WMMA_F16(ap, load_b_frag(Vp, SEQ,  0, kb, lane), o0.v);
    o1.v = WMMA_F16(ap, load_b_frag(Vp, SEQ, 16, kb, lane), o1.v);
    o2.v = WMMA_F16(ap, load_b_frag(Vp, SEQ, 32, kb, lane), o2.v);
    o3.v = WMMA_F16(ap, load_b_frag(Vp, SEQ, 48, kb, lane), o3.v);
  }
  #pragma unroll
  for (int r = 0; r < 8; r++) {
    float inv = 1.0f / l[r];
    int row = q0 + rbase + r;
    size_t base = ((size_t)(bi * SEQ + row)) * DIM + hi * HD;
    AO[base +  0 + colL] = (_Float16)(o0.f[r] * inv);
    AO[base + 16 + colL] = (_Float16)(o1.f[r] * inv);
    AO[base + 32 + colL] = (_Float16)(o2.f[r] * inv);
    AO[base + 48 + colL] = (_Float16)(o3.f[r] * inv);
  }
}

// ---- output projection + final transpose: out(b,d,s) = (AO @ W2)^T ------
// Wave tile 32 x 64 (2x4 frags), direct global loads (everything is L2-resident).
__global__ void __launch_bounds__(128) k_out(const _Float16* __restrict__ AO,
                                             const _Float16* __restrict__ W2t,
                                             float* __restrict__ out) {
  int lane = threadIdx.x & 31;
  int wid  = threadIdx.x >> 5;
  int tile = blockIdx.x * 4 + wid;                   // 128 row-tiles x 16 col-tiles
  int rt = tile >> 4;
  int ct = tile & 15;
  int g0r = rt * 32, n0 = ct * 64;
  v8f c[2][4];
  #pragma unroll
  for (int i = 0; i < 2; i++)
    #pragma unroll
    for (int j = 0; j < 4; j++) c[i][j] = (v8f){};
  #pragma unroll 2
  for (int k0 = 0; k0 < DIM; k0 += 32) {
    __builtin_prefetch(AO + (size_t)(g0r + (lane & 15)) * DIM + k0 + 64, 0, 1);
    v16h a0 = load_a_frag(AO, DIM, g0r,      k0, lane);
    v16h a1 = load_a_frag(AO, DIM, g0r + 16, k0, lane);
    #pragma unroll
    for (int j = 0; j < 4; j++) {
      v16h b = load_b_frag(W2t, DIM, n0 + j * 16, k0, lane);
      c[0][j] = WMMA_F16(a0, b, c[0][j]);
      c[1][j] = WMMA_F16(a1, b, c[1][j]);
    }
  }
  int rbase = (lane >> 4) << 3, colL = lane & 15;
  #pragma unroll
  for (int rf = 0; rf < 2; rf++) {
    #pragma unroll
    for (int j = 0; j < 4; j++) {
      F8 cc; cc.v = c[rf][j];
      int n = n0 + j * 16 + colL;
      #pragma unroll
      for (int r = 0; r < 8; r++) {
        int g = g0r + rf * 16 + rbase + r;
        int bi = g >> 11, sidx = g & (SEQ - 1);
        out[((size_t)bi * DIM + n) * SEQ + sidx] = cc.f[r];
      }
    }
  }
}

extern "C" void kernel_launch(void* const* d_in, const int* in_sizes, int n_in,
                              void* d_out, int out_size, void* d_ws, size_t ws_size,
                              hipStream_t stream) {
  (void)in_sizes; (void)n_in; (void)out_size; (void)ws_size;
  const float* x  = (const float*)d_in[0];
  const float* W1 = (const float*)d_in[1];
  const float* W2 = (const float*)d_in[2];
  float* out = (float*)d_out;

  char* ws = (char*)d_ws;
  const size_t XH_B  = (size_t)BDIM * SEQ * DIM * 2;           // 8 MB
  const size_t W1T_B = (size_t)TD * DIM * 2;                   // 6 MB
  const size_t W2T_B = (size_t)DIM * DIM * 2;                  // 2 MB
  const size_t QKV_B = (size_t)BDIM * NH * SEQ * HD * 2;       // 8 MB each
  _Float16* Xh  = (_Float16*)(ws);
  _Float16* W1t = (_Float16*)(ws + XH_B);
  _Float16* W2t = (_Float16*)(ws + XH_B + W1T_B);
  _Float16* Q   = (_Float16*)(ws + XH_B + W1T_B + W2T_B);
  _Float16* K   = (_Float16*)(ws + XH_B + W1T_B + W2T_B + QKV_B);
  _Float16* Vt  = (_Float16*)(ws + XH_B + W1T_B + W2T_B + 2 * QKV_B);
  _Float16* AO  = (_Float16*)(ws + XH_B + W1T_B + W2T_B + 3 * QKV_B);

  k_transpose_x<<<(BDIM * SEQ * DIM) / 256, 256, 0, stream>>>(x, Xh);
  k_w1t<<<(TD * DIM) / 256, 256, 0, stream>>>(W1, W1t);
  k_w2t<<<(DIM * DIM) / 256, 256, 0, stream>>>(W2, W2t);

  // QKV: 128 row-tiles (32 rows) x 12 col-tiles (256 cols) = 1536 blocks
  k_qkv<<<128 * (TD / 256), 128, 0, stream>>>(Xh, W1t, Q, K, Vt);
  // attention: B*NH*(SEQ/16) = 4096 waves, 4 per block
  k_attn<<<4096 / 4, 128, 0, stream>>>(Q, K, Vt, AO);
  // output proj: 128 row-tiles x 16 col-tiles = 2048 waves, 4 per block
  k_out<<<2048 / 4, 128, 0, stream>>>(AO, W2t, out);
}
